// LightweightGraphEncoder_19172734009947
// MI455X (gfx1250) — matrix operand order repaired and checked by reading, not compiled, not run on hardware
//
#include <hip/hip_runtime.h>
#include <hip/hip_bf16.h>
#include <math.h>

typedef __attribute__((ext_vector_type(16))) _Float16 v16h;
typedef __attribute__((ext_vector_type(8)))  float    v8f;

#define LN_EPS 1e-5f

// ---------------------------------------------------------------------------
// K0: h = normalize(x @ Wp + bp)  ->  f16 [N,16] for WMMA similarity
// ---------------------------------------------------------------------------
__global__ void k_project(const float* __restrict__ x, const float* __restrict__ Wp,
                          const float* __restrict__ bp, _Float16* __restrict__ hn,
                          int N, int D) {
  int i = blockIdx.x * blockDim.x + threadIdx.x;
  if (i >= N) return;
  float h[16];
  for (int c = 0; c < 16; ++c) {
    float acc = bp[c];
    for (int k = 0; k < D; ++k) acc += x[i * D + k] * Wp[k * 16 + c];
    h[c] = acc;
  }
  float nrm = 0.f;
  for (int c = 0; c < 16; ++c) nrm += h[c] * h[c];
  float inv = 1.0f / fmaxf(sqrtf(nrm), 1e-12f);
  for (int c = 0; c < 16; ++c) hn[i * 16 + c] = (_Float16)(h[c] * inv);
}

// ---------------------------------------------------------------------------
// K1: fused S = h h^T tile GEMM (WMMA f16 16x16x32, K padded 16->32)
//     + streaming per-row top-6 (self excluded).
//     Block = 256 threads (8 waves); block owns 16 rows. Wave w sweeps column
//     tiles {4w..4w+3} + k*32 (4 WMMAs per iteration), fully wave-local:
//     per-wave LDS tile + s_wait_dscnt instead of block barriers. One
//     __syncthreads only at the final 8-way candidate merge.
// ---------------------------------------------------------------------------
__global__ void __launch_bounds__(256) k_topk(const _Float16* __restrict__ hn,
                                              int* __restrict__ nidx,
                                              float* __restrict__ nval, int N) {
  __shared__ float s_tile[8][4][16][16];
  __shared__ float s_cval[8][16][6];
  __shared__ int   s_cidx[8][16][6];

  const int wave = threadIdx.x >> 5;
  const int lane = threadIdx.x & 31;
  const int rowBase = blockIdx.x * 16;

  // A operand: 16x32 f16. lanes0-15: K0..15 in halves0..15; lanes16-31: zero
  // (the zero-padded K=16..31 block).
  v16h a;
  {
    const int r = rowBase + (lane & 15);
    for (int e = 0; e < 16; ++e) a[e] = (lane < 16) ? hn[r * 16 + e] : (_Float16)0.0f;
  }

  float tv[6]; int ti[6];
  for (int s = 0; s < 6; ++s) { tv[s] = -3.0e38f; ti[s] = -1; }

  const int nColTiles = N >> 4;                       // 512
  for (int ct = wave * 4; ct < nColTiles; ct += 32) { // 16 iterations/wave
    // B operands: 32x16 f16. lanes0-15: col n=lane, K0..15; lanes16-31: zero.
    v16h b[4];
    if (lane < 16) {
      for (int t = 0; t < 4; ++t) {
        const int cidx = (ct + t) * 16 + lane;
        for (int e = 0; e < 16; ++e) b[t][e] = hn[cidx * 16 + e];
      }
      if (ct + 32 < nColTiles)  // pull next stripe toward the WGP
        __builtin_prefetch((const void*)(hn + (size_t)(ct + 32) * 256 + lane * 16), 0, 3);
    } else {
      for (int t = 0; t < 4; ++t)
        for (int e = 0; e < 16; ++e) b[t][e] = (_Float16)0.0f;
    }

    // 4 back-to-back WMMAs (EXEC all-ones here; guard above has rejoined)
    v8f c0 = {}, c1 = {}, c2 = {}, c3 = {};
    c0 = __builtin_amdgcn_wmma_f32_16x16x32_f16(false, a, false, b[0], (short)0, c0, false, false);
    c1 = __builtin_amdgcn_wmma_f32_16x16x32_f16(false, a, false, b[1], (short)0, c1, false, false);
    c2 = __builtin_amdgcn_wmma_f32_16x16x32_f16(false, a, false, b[2], (short)0, c2, false, false);
    c3 = __builtin_amdgcn_wmma_f32_16x16x32_f16(false, a, false, b[3], (short)0, c3, false, false);

    // C/D layout: VGPR v, lane l -> (M = v + 8*(l>=16), N = l&15)
    {
      const int moff = (lane < 16) ? 0 : 8;
      const int ncol = lane & 15;
      for (int v = 0; v < 8; ++v) {
        s_tile[wave][0][moff + v][ncol] = c0[v];
        s_tile[wave][1][moff + v][ncol] = c1[v];
        s_tile[wave][2][moff + v][ncol] = c2[v];
        s_tile[wave][3][moff + v][ncol] = c3[v];
      }
    }
    // wave-local cross-lane handoff: DS ops are in-order per wave, so a
    // dscnt wait makes all lanes' tile stores visible to this wave's loads.
    asm volatile("s_wait_dscnt 0x0" ::: "memory");

    if (lane < 16) {
      const int gr = rowBase + lane;
      for (int t = 0; t < 4; ++t) {
        for (int cc = 0; cc < 16; ++cc) {
          const int gj = (ct + t) * 16 + cc;
          const float v = s_tile[wave][t][lane][cc];
          if (gj != gr && v > tv[5]) {
            int p = 5;
            while (p > 0 && v > tv[p - 1]) { tv[p] = tv[p - 1]; ti[p] = ti[p - 1]; --p; }
            tv[p] = v; ti[p] = gj;
          }
        }
      }
    }
    // hardware keeps next iteration's DS stores ordered behind these loads;
    // the asm memory clobber at loop top blocks compiler reordering.
  }

  if (lane < 16)
    for (int s = 0; s < 6; ++s) { s_cval[wave][lane][s] = tv[s]; s_cidx[wave][lane][s] = ti[s]; }
  __syncthreads();

  // merge the 8 per-wave candidate lists (48 -> 6) per row
  if (threadIdx.x < 16) {
    const int r = threadIdx.x;
    float bv[6]; int bi[6];
    for (int s = 0; s < 6; ++s) { bv[s] = -3.0e38f; bi[s] = -1; }
    for (int w = 0; w < 8; ++w)
      for (int s = 0; s < 6; ++s) {
        const float v = s_cval[w][r][s]; const int j = s_cidx[w][r][s];
        if (j >= 0 && v > bv[5]) {
          int p = 5;
          while (p > 0 && v > bv[p - 1]) { bv[p] = bv[p - 1]; bi[p] = bi[p - 1]; --p; }
          bv[p] = v; bi[p] = j;
        }
      }
    const int gr = rowBase + r;
    for (int s = 0; s < 6; ++s) { nval[gr * 6 + s] = bv[s]; nidx[gr * 6 + s] = bi[s]; }
  }
}

// ---------------------------------------------------------------------------
// graph normalization scalars
// ---------------------------------------------------------------------------
__global__ void k_zero(float* __restrict__ p, int n) {
  int i = blockIdx.x * blockDim.x + threadIdx.x;
  if (i < n) p[i] = 0.f;
}

// symmetric row sums: each directed edge (i,j,v) contributes v/2 to both rows
__global__ void k_edges_rs(const int* __restrict__ nidx, const float* __restrict__ nval,
                           float* __restrict__ rs, int N) {
  int e = blockIdx.x * blockDim.x + threadIdx.x;
  if (e >= N * 6) return;
  int i = e / 6;
  int j = nidx[e];
  if (j < 0) return;
  float v = 0.5f * nval[e];
  atomicAdd(&rs[i], v);
  atomicAdd(&rs[j], v);
}

// d_i = (A.sum(1))^-0.5 where A.sum(1) = 1 + rs/(rs+1e-8)   (adj row-normalized, +I)
__global__ void k_degree(const float* __restrict__ rs, float* __restrict__ dd, int N) {
  int i = blockIdx.x * blockDim.x + threadIdx.x;
  if (i >= N) return;
  float asum = 1.0f + rs[i] / (rs[i] + 1e-8f);
  dd[i] = rsqrtf(fmaxf(asum, 1e-12f));
}

// ---------------------------------------------------------------------------
// per-layer dense work: hw = h@W ; mh = relu(h@w0 + b0)
// ---------------------------------------------------------------------------
__global__ void k_dense(const float* __restrict__ h, int N, int Din,
                        const float* __restrict__ W, int Hout, float* __restrict__ hw,
                        const float* __restrict__ w0, int Hmid,
                        const float* __restrict__ b0, float* __restrict__ mh) {
  int tid = blockIdx.x * blockDim.x + threadIdx.x;
  int tot = Hout + Hmid;
  if (tid >= N * tot) return;
  int i = tid / tot, c = tid % tot;
  const float* hr = h + (size_t)i * Din;
  if (c < Hout) {
    float acc = 0.f;
    for (int k = 0; k < Din; ++k) acc += hr[k] * W[k * Hout + c];
    hw[(size_t)i * Hout + c] = acc;
  } else {
    int c2 = c - Hout;
    float acc = b0[c2];
    for (int k = 0; k < Din; ++k) acc += hr[k] * w0[k * Hmid + c2];
    mh[(size_t)i * Hmid + c2] = fmaxf(acc, 0.f);
  }
}

// agg init: diagonal of norm_adj (+I term) plus bias
__global__ void k_agg_init(const float* __restrict__ hw, const float* __restrict__ dd,
                           const float* __restrict__ gb, int Hout,
                           float* __restrict__ agg, int N) {
  int tid = blockIdx.x * blockDim.x + threadIdx.x;
  if (tid >= N * Hout) return;
  int i = tid / Hout, c = tid % Hout;
  float d = dd[i];
  agg[tid] = d * d * hw[(size_t)i * Hout + c] + gb[c];
}

// sparse aggregation: scatter each half-edge into both endpoint rows
__global__ void k_scatter(const int* __restrict__ nidx, const float* __restrict__ nval,
                          const float* __restrict__ hw, const float* __restrict__ dd,
                          const float* __restrict__ rs, int Hout,
                          float* __restrict__ agg, int N) {
  int tid = blockIdx.x * blockDim.x + threadIdx.x;
  if (tid >= N * 6 * Hout) return;
  int e = tid / Hout, c = tid % Hout;
  int i = e / 6;
  int j = nidx[e];
  if (j < 0) return;
  float v  = 0.5f * nval[e];
  float dij = dd[i] * dd[j] * v;
  atomicAdd(&agg[(size_t)i * Hout + c], dij / (rs[i] + 1e-8f) * hw[(size_t)j * Hout + c]);
  atomicAdd(&agg[(size_t)j * Hout + c], dij / (rs[j] + 1e-8f) * hw[(size_t)i * Hout + c]);
}

// layernorm stats per row
__global__ void k_lnstats(const float* __restrict__ agg, int Hout,
                          float* __restrict__ mean, float* __restrict__ rstd, int N) {
  int i = blockIdx.x * blockDim.x + threadIdx.x;
  if (i >= N) return;
  const float* r = agg + (size_t)i * Hout;
  float m = 0.f;
  for (int c = 0; c < Hout; ++c) m += r[c];
  m /= (float)Hout;
  float v = 0.f;
  for (int c = 0; c < Hout; ++c) { float d = r[c] - m; v += d * d; }
  v /= (float)Hout;
  mean[i] = m;
  rstd[i] = rsqrtf(v + LN_EPS);
}

// h_next = relu(LN(agg)*g + bt) + mh @ w1 + b1
__global__ void k_final(const float* __restrict__ agg, const float* __restrict__ mean,
                        const float* __restrict__ rstd, const float* __restrict__ g,
                        const float* __restrict__ bt, const float* __restrict__ mh,
                        const float* __restrict__ w1, int Hmid,
                        const float* __restrict__ b1, int Hout,
                        float* __restrict__ hout, int N) {
  int tid = blockIdx.x * blockDim.x + threadIdx.x;
  if (tid >= N * Hout) return;
  int i = tid / Hout, c = tid % Hout;
  float ln = (agg[tid] - mean[i]) * rstd[i] * g[c] + bt[c];
  ln = fmaxf(ln, 0.f);
  float acc = b1[c];
  const float* mr = mh + (size_t)i * Hmid;
  for (int k = 0; k < Hmid; ++k) acc += mr[k] * w1[k * Hout + c];
  hout[tid] = ln + acc;
}

// ---------------------------------------------------------------------------
extern "C" void kernel_launch(void* const* d_in, const int* in_sizes, int n_in,
                              void* d_out, int out_size, void* d_ws, size_t ws_size,
                              hipStream_t stream) {
  const float* x   = (const float*)d_in[0];
  const float* Wp  = (const float*)d_in[1];
  const float* bp  = (const float*)d_in[2];
  const float* gw0 = (const float*)d_in[3];  const float* gb0 = (const float*)d_in[4];
  const float* gw1 = (const float*)d_in[5];  const float* gb1 = (const float*)d_in[6];
  const float* gw2 = (const float*)d_in[7];  const float* gb2 = (const float*)d_in[8];
  const float* lg0 = (const float*)d_in[9];  const float* lb0 = (const float*)d_in[10];
  const float* lg1 = (const float*)d_in[11]; const float* lb1 = (const float*)d_in[12];
  const float* lg2 = (const float*)d_in[13]; const float* lb2 = (const float*)d_in[14];
  const float* riw0= (const float*)d_in[15]; const float* rib0= (const float*)d_in[16];
  const float* riw1= (const float*)d_in[17]; const float* rib1= (const float*)d_in[18];
  const float* rhw0= (const float*)d_in[19]; const float* rhb0= (const float*)d_in[20];
  const float* rhw1= (const float*)d_in[21]; const float* rhb1= (const float*)d_in[22];
  const float* row0= (const float*)d_in[23]; const float* rob0= (const float*)d_in[24];
  const float* row1= (const float*)d_in[25]; const float* rob1= (const float*)d_in[26];
  (void)n_in; (void)out_size; (void)ws_size;

  const int D = 20, H = 32, O = 128;
  const int N = in_sizes[0] / D;   // 8192

  // workspace carve (256B aligned)
  char* p = (char*)d_ws;
  auto carve = [&](size_t bytes) -> void* {
    void* r = (void*)p;
    p += (bytes + 255) & ~(size_t)255;
    return r;
  };
  _Float16* hn  = (_Float16*)carve((size_t)N * 16 * sizeof(_Float16));
  int*      nidx= (int*)     carve((size_t)N * 6 * sizeof(int));
  float*    nval= (float*)   carve((size_t)N * 6 * sizeof(float));
  float*    rs  = (float*)   carve((size_t)N * sizeof(float));
  float*    dd  = (float*)   carve((size_t)N * sizeof(float));
  float*    hw  = (float*)   carve((size_t)N * O * sizeof(float));
  float*    mh  = (float*)   carve((size_t)N * O * sizeof(float));
  float*    agg = (float*)   carve((size_t)N * O * sizeof(float));
  float*    mean= (float*)   carve((size_t)N * sizeof(float));
  float*    rstd= (float*)   carve((size_t)N * sizeof(float));
  float*    hA  = (float*)   carve((size_t)N * O * sizeof(float));
  float*    hB  = (float*)   carve((size_t)N * O * sizeof(float));

  const int BT = 256;
  k_project<<<(N + BT - 1) / BT, BT, 0, stream>>>(x, Wp, bp, hn, N, D);
  k_topk<<<N / 16, 256, 0, stream>>>(hn, nidx, nval, N);
  k_zero<<<(N + BT - 1) / BT, BT, 0, stream>>>(rs, N);
  k_edges_rs<<<(6 * N + BT - 1) / BT, BT, 0, stream>>>(nidx, nval, rs, N);
  k_degree<<<(N + BT - 1) / BT, BT, 0, stream>>>(rs, dd, N);

  struct Layer {
    const float* h; int Din; const float* W; const float* gb; int Hout;
    const float* w0; int Hmid; const float* b0; const float* w1; const float* b1;
    const float* g; const float* bt; float* out;
  };
  Layer L[3] = {
    { x,  D, gw0, gb0, H, riw0, H, rib0, riw1, rib1, lg0, lb0, hA },
    { hA, H, gw1, gb1, H, rhw0, H, rhb0, rhw1, rhb1, lg1, lb1, hB },
    { hB, H, gw2, gb2, O, row0, O, rob0, row1, rob1, lg2, lb2, (float*)d_out },
  };

  for (int l = 0; l < 3; ++l) {
    const Layer& q = L[l];
    int totDense = N * (q.Hout + q.Hmid);
    k_dense<<<(totDense + BT - 1) / BT, BT, 0, stream>>>(
        q.h, N, q.Din, q.W, q.Hout, hw, q.w0, q.Hmid, q.b0, mh);
    k_agg_init<<<(N * q.Hout + BT - 1) / BT, BT, 0, stream>>>(hw, dd, q.gb, q.Hout, agg, N);
    k_scatter<<<(N * 6 * q.Hout + BT - 1) / BT, BT, 0, stream>>>(
        nidx, nval, hw, dd, rs, q.Hout, agg, N);
    k_lnstats<<<(N + BT - 1) / BT, BT, 0, stream>>>(agg, q.Hout, mean, rstd, N);
    k_final<<<(N * q.Hout + BT - 1) / BT, BT, 0, stream>>>(
        agg, mean, rstd, q.g, q.bt, mh, q.w1, q.Hmid, q.b1, q.Hout, q.out, N);
  }
}
